// S2ARoIHead_18047452578535
// MI455X (gfx1250) — matrix operand order repaired and checked by reading, not compiled, not run on hardware
//
#include <hip/hip_runtime.h>
#include <hip/hip_bf16.h>

// ---------------------------------------------------------------------------
// Types for CDNA5 WMMA (wave32). 16x16x32 bf16 -> f32 accumulate.
// ---------------------------------------------------------------------------
typedef __bf16 bf16_t;
typedef __attribute__((ext_vector_type(16))) __bf16 v16bf;
typedef __attribute__((ext_vector_type(8)))  __bf16 v8bf;
typedef __attribute__((ext_vector_type(8)))  float  v8f;

#define WMMA_BF16(A_, B_, C_) \
  __builtin_amdgcn_wmma_f32_16x16x32_bf16(false, (A_), false, (B_), (short)0, (C_), false, false)

#define NPROP   2048
#define CCH     256
#define FH      64
#define FW      64
#define P49     49
#define FD      12544      // 256*49
#define KCONV   2304       // 9*256  (k = s*256 + ci, s = ky*3+kx)
#define HID     1024
#define NCLS    16
#define CLAMPV  4.135f
#define SCALEV  0.0625f    // 1/16

union AFrag { v16bf v; v8bf h[2]; };

// ---------------------------------------------------------------------------
// 0a) fp32 -> bf16 (plain)
// ---------------------------------------------------------------------------
__global__ void cvt_f32_bf16(const float* __restrict__ s, bf16_t* __restrict__ d, int n) {
  int i = blockIdx.x * blockDim.x + threadIdx.x;
  if (i < n) d[i] = (bf16_t)s[i];
}

// 0b) conv_w [O][I][3][3] fp32 -> bf16 [O][s*256+ci] (spatial-major K)
__global__ void cvt_convw(const float* __restrict__ s, bf16_t* __restrict__ d) {
  int i = blockIdx.x * blockDim.x + threadIdx.x;
  if (i < CCH * KCONV) {
    int o = i / KCONV, k = i - o * KCONV;
    int sp = k >> 8, ci = k & 255;
    d[i] = (bf16_t)s[(o * CCH + ci) * 9 + sp];
  }
}

// ---------------------------------------------------------------------------
// 1) RoIAlign (sampling_ratio=1) -> box_feat bf16 [n][c][p], p = y*7+x
// ---------------------------------------------------------------------------
__global__ __launch_bounds__(256) void roi_align_k(const float* __restrict__ feat,
                                                   const float* __restrict__ boxes,
                                                   const int*   __restrict__ bidx,
                                                   bf16_t*      __restrict__ bfeat) {
  int n = blockIdx.x;
  __shared__ int   sx0[P49], sy0[P49], sx1[P49], sy1[P49];
  __shared__ float sw00[P49], sw01[P49], sw10[P49], sw11[P49];
  int t = threadIdx.x;
  if (t < P49) {
    float x1 = boxes[n*4+0]*SCALEV, y1 = boxes[n*4+1]*SCALEV;
    float x2 = boxes[n*4+2]*SCALEV, y2 = boxes[n*4+3]*SCALEV;
    float bw = fmaxf(x2 - x1, 1.f) * (1.f/7.f);
    float bh = fmaxf(y2 - y1, 1.f) * (1.f/7.f);
    int py = t / 7, px = t - 7*py;
    float ys = fminf(fmaxf(y1 + ((float)py + 0.5f) * bh, 0.f), 63.f);
    float xs = fminf(fmaxf(x1 + ((float)px + 0.5f) * bw, 0.f), 63.f);
    float y0 = floorf(ys), x0 = floorf(xs);
    int y0i = (int)y0, x0i = (int)x0;
    float wy1 = ys - y0, wx1 = xs - x0, wy0 = 1.f - wy1, wx0 = 1.f - wx1;
    sx0[t] = x0i; sy0[t] = y0i;
    sx1[t] = min(x0i + 1, 63); sy1[t] = min(y0i + 1, 63);
    sw00[t] = wy0*wx0; sw01[t] = wy0*wx1; sw10[t] = wy1*wx0; sw11[t] = wy1*wx1;
  }
  __syncthreads();
  int b = bidx[n];
  for (int idx = t; idx < FD; idx += 256) {
    int c = idx / P49, p = idx - P49 * c;
    const float* f = feat + ((size_t)(b * CCH + c)) * (FH * FW);
    float v = sw00[p] * f[sy0[p]*FW + sx0[p]]
            + sw01[p] * f[sy0[p]*FW + sx1[p]]
            + sw10[p] * f[sy1[p]*FW + sx0[p]]
            + sw11[p] * f[sy1[p]*FW + sx1[p]];
    bfeat[(size_t)n * FD + idx] = (bf16_t)v;
  }
}

// ---------------------------------------------------------------------------
// 2) Fused 3x3 conv (implicit GEMM, WMMA bf16) + ReLU + 5-dim coarse proj.
//    One WG (512 thr = 16 waves) per proposal. box_feat staged in LDS as
//    spatial-major/channel-minor zero-padded 9x9 -> A-fragment = 2x
//    ds_load_b128. B = permuted bf16 conv weights (contiguous 32B loads).
//    Ping-pong (unroll-2) K loop: no buffer-rotation register copies.
// ---------------------------------------------------------------------------
__global__ __launch_bounds__(512) void conv_coarse_k(const bf16_t* __restrict__ bfeat,
                                                     const bf16_t* __restrict__ wconv, // [256][2304]
                                                     const float*  __restrict__ convb,
                                                     const float*  __restrict__ alignw, // [5][12544]
                                                     const float*  __restrict__ alignb,
                                                     float*        __restrict__ coarse) {
  int n = blockIdx.x;
  __shared__ bf16_t sfeat[81 * CCH];   // [(yy*9+xx)][c], zero-padded border
  __shared__ float  sc[5];
  int t = threadIdx.x;
  for (int i = t; i < 81 * CCH; i += 512) sfeat[i] = (bf16_t)0.f;
  if (t < 5) sc[t] = 0.f;
  __syncthreads();
  for (int i = t; i < FD; i += 512) {
    int c = i / P49, p = i - P49 * c, y = p / 7, x = p - 7 * y;
    sfeat[((y + 1) * 9 + (x + 1)) * CCH + c] = bfeat[(size_t)n * FD + i];
  }
  __syncthreads();

  int lane   = t & 31;
  int w      = t >> 5;        // 0..15
  int mtile  = w & 3;         // rows 0..63 (rows >=49 are garbage, discarded)
  int ngroup = w >> 2;        // 4 groups of 64 output channels
  int r      = lane & 15;
  int khalf  = lane >> 4;
  int p      = mtile * 16 + r;
  int pc     = min(p, P49 - 1);        // clamp: garbage rows harmless
  int y = pc / 7, x = pc - 7 * y;

  v8f zero8 = {0.f,0.f,0.f,0.f,0.f,0.f,0.f,0.f};
  v8f acc[4]; acc[0] = zero8; acc[1] = zero8; acc[2] = zero8; acc[3] = zero8;

  const bf16_t* wb0 = wconv + (size_t)(ngroup * 64 + r) * KCONV + khalf * 16;

  auto loadA = [&](int kc, AFrag& af) {
    int s  = kc >> 3;                   // 0..8
    int s3 = (s * 11) >> 5;             // s/3 for s<=8
    int sm = s - 3 * s3;
    int spat = (y + s3) * 9 + (x + sm);
    const bf16_t* base = sfeat + spat * CCH + (kc & 7) * 32 + khalf * 16;
    af.h[0] = *(const v8bf*)base;
    af.h[1] = *(const v8bf*)(base + 8);
  };
  auto loadB = [&](int kc, v16bf* bfr) {
    #pragma unroll
    for (int tt = 0; tt < 4; ++tt)
      bfr[tt] = *(const v16bf*)(wb0 + (size_t)tt * 16 * KCONV + kc * 32);
  };
  auto domma = [&](AFrag& af, v16bf* bfr) {
    #pragma unroll
    for (int tt = 0; tt < 4; ++tt) acc[tt] = WMMA_BF16(af.v, bfr[tt], acc[tt]);
  };

  AFrag a0, a1; v16bf b0[4], b1[4];
  loadA(0, a0); loadB(0, b0);
  int kc = 0;
  for (; kc + 2 < KCONV / 32; kc += 2) {
    loadA(kc + 1, a1); loadB(kc + 1, b1);
    domma(a0, b0);
    loadA(kc + 2, a0); loadB(kc + 2, b0);
    domma(a1, b1);
  }
  loadA(kc + 1, a1); loadB(kc + 1, b1);
  domma(a0, b0);
  domma(a1, b1);

  float part[5] = {0.f, 0.f, 0.f, 0.f, 0.f};
  #pragma unroll
  for (int tt = 0; tt < 4; ++tt) {
    int c = ngroup * 64 + tt * 16 + r;
    float cb = convb[c];
    #pragma unroll
    for (int v = 0; v < 8; ++v) {
      int pr = mtile * 16 + v + khalf * 8;
      float valid = (pr < P49) ? 1.f : 0.f;
      int prc = min(pr, P49 - 1);
      float s = fmaxf(acc[tt][v] + cb, 0.f) * valid;
      #pragma unroll
      for (int j = 0; j < 5; ++j)
        part[j] += s * alignw[j * FD + c * P49 + prc];
    }
  }
  #pragma unroll
  for (int j = 0; j < 5; ++j) atomicAdd(&sc[j], part[j]);
  __syncthreads();
  if (t < 5) {
    float v = sc[t] + alignb[t];
    coarse[n * 5 + t] = fminf(fmaxf(v, -CLAMPV), CLAMPV);
  }
}

// ---------------------------------------------------------------------------
// 3) Decode deltas -> rotated grid -> grid_sample(zeros) -> aligned bf16
// ---------------------------------------------------------------------------
__global__ __launch_bounds__(256) void grid_sample_k(const bf16_t* __restrict__ bfeat,
                                                     const float*  __restrict__ boxes,
                                                     const float*  __restrict__ coarse,
                                                     bf16_t*       __restrict__ alg) {
  int n = blockIdx.x;
  __shared__ int   sx0[P49], sy0[P49];
  __shared__ float sw00[P49], sw01[P49], sw10[P49], sw11[P49];
  int t = threadIdx.x;
  if (t < P49) {
    float p0 = boxes[n*4+0], p1 = boxes[n*4+1], p2 = boxes[n*4+2], p3 = boxes[n*4+3];
    float w_ = p2 - p0, h_ = p3 - p1;
    float dw = coarse[n*5+2], dh = coarse[n*5+3], da = coarse[n*5+4];
    float w2 = w_ * __expf(dw), h2 = h_ * __expf(dh);
    float cth = __cosf(da), sth = __sinf(da);
    float sxv = w2 * (1.f/7.f), syv = h2 * (1.f/7.f);
    int py = t / 7, px = t - 7 * py;
    float xb = (2.f * (float)px + 1.f) * (1.f/7.f) - 1.f;
    float yb = (2.f * (float)py + 1.f) * (1.f/7.f) - 1.f;
    float gx = sxv * cth * xb - syv * sth * yb;
    float gy = sxv * sth * xb + syv * cth * yb;
    float ix = ((gx + 1.f) * 7.f - 1.f) * 0.5f;
    float iy = ((gy + 1.f) * 7.f - 1.f) * 0.5f;
    float x0 = floorf(ix), y0 = floorf(iy);
    int x0i = (int)x0, y0i = (int)y0;
    float wx1 = ix - x0, wy1 = iy - y0, wx0 = 1.f - wx1, wy0 = 1.f - wy1;
    float v00 = (y0i   >= 0 && y0i   < 7 && x0i   >= 0 && x0i   < 7) ? 1.f : 0.f;
    float v01 = (y0i   >= 0 && y0i   < 7 && x0i+1 >= 0 && x0i+1 < 7) ? 1.f : 0.f;
    float v10 = (y0i+1 >= 0 && y0i+1 < 7 && x0i   >= 0 && x0i   < 7) ? 1.f : 0.f;
    float v11 = (y0i+1 >= 0 && y0i+1 < 7 && x0i+1 >= 0 && x0i+1 < 7) ? 1.f : 0.f;
    sx0[t] = x0i; sy0[t] = y0i;
    sw00[t] = wy0*wx0*v00; sw01[t] = wy0*wx1*v01; sw10[t] = wy1*wx0*v10; sw11[t] = wy1*wx1*v11;
  }
  __syncthreads();
  const bf16_t* fb = bfeat + (size_t)n * FD;
  for (int idx = t; idx < FD; idx += 256) {
    int c = idx / P49, p = idx - P49 * c;
    int x0i = sx0[p], y0i = sy0[p];
    int xc0 = min(max(x0i, 0), 6),     yc0 = min(max(y0i, 0), 6);
    int xc1 = min(max(x0i + 1, 0), 6), yc1 = min(max(y0i + 1, 0), 6);
    const bf16_t* f = fb + c * P49;
    float v = sw00[p] * (float)f[yc0*7 + xc0] + sw01[p] * (float)f[yc0*7 + xc1]
            + sw10[p] * (float)f[yc1*7 + xc0] + sw11[p] * (float)f[yc1*7 + xc1];
    alg[(size_t)n * FD + idx] = (bf16_t)v;
  }
}

// ---------------------------------------------------------------------------
// 4) bf16 WMMA GEMM: C = relu(A[M][K] @ B[Nout][K]^T + bias).
//    Block 256 thr (8 waves), tile 64(M) x 128(N). Ping-pong K loop.
//    Requires M%64==0, Nout%128==0, K%64==0 (fc1: 12544, fc2: 1024).
// ---------------------------------------------------------------------------
__global__ __launch_bounds__(256) void gemm_bf16_k(const bf16_t* __restrict__ A,
                                                   const bf16_t* __restrict__ B,
                                                   const float*  __restrict__ bias,
                                                   bf16_t*       __restrict__ C,
                                                   int M, int K, int Nout, int relu) {
  int lane  = threadIdx.x & 31;
  int w     = threadIdx.x >> 5;
  int mbase = blockIdx.y * 64  + (w & 3) * 16;
  int nbase = blockIdx.x * 128 + (w >> 2) * 64;
  int r     = lane & 15;
  int khalf = lane >> 4;

  const bf16_t* arow = A + (size_t)(mbase + r) * K + khalf * 16;
  const bf16_t* brow = B + (size_t)(nbase + r) * K + khalf * 16;

  v8f zero8 = {0.f,0.f,0.f,0.f,0.f,0.f,0.f,0.f};
  v8f acc[4]; acc[0] = zero8; acc[1] = zero8; acc[2] = zero8; acc[3] = zero8;

  auto loadA = [&](int k0, AFrag& af) {
    af.h[0] = *(const v8bf*)(arow + k0);
    af.h[1] = *(const v8bf*)(arow + k0 + 8);
  };
  auto loadB = [&](int k0, v16bf* bfr) {
    #pragma unroll
    for (int tt = 0; tt < 4; ++tt)
      bfr[tt] = *(const v16bf*)(brow + (size_t)tt * 16 * K + k0);
  };
  auto domma = [&](AFrag& af, v16bf* bfr) {
    #pragma unroll
    for (int tt = 0; tt < 4; ++tt) acc[tt] = WMMA_BF16(af.v, bfr[tt], acc[tt]);
  };

  AFrag a0, a1; v16bf b0[4], b1[4];
  loadA(0, a0); loadB(0, b0);
  int k0 = 0;
  for (; k0 + 64 < K; k0 += 64) {
    __builtin_prefetch(arow + k0 + 1024, 0, 0);   // global_prefetch_b8 ahead
    loadA(k0 + 32, a1); loadB(k0 + 32, b1);
    domma(a0, b0);
    loadA(k0 + 64, a0); loadB(k0 + 64, b0);
    domma(a1, b1);
  }
  loadA(k0 + 32, a1); loadB(k0 + 32, b1);
  domma(a0, b0);
  domma(a1, b1);

  #pragma unroll
  for (int tt = 0; tt < 4; ++tt) {
    int col = nbase + tt * 16 + r;
    float bb = bias ? bias[col] : 0.f;
    #pragma unroll
    for (int v = 0; v < 8; ++v) {
      int row = mbase + v + khalf * 8;
      float s = acc[tt][v] + bb;
      if (relu) s = fmaxf(s, 0.f);
      C[(size_t)row * Nout + col] = (bf16_t)s;
    }
  }
}

// ---------------------------------------------------------------------------
// 5) Heads: logits/box dots + softmax + rbox decode -> out [N][21] fp32
// ---------------------------------------------------------------------------
__global__ __launch_bounds__(32) void heads_k(const bf16_t* __restrict__ r2,
                                              const float* __restrict__ clsw, const float* __restrict__ clsb,
                                              const float* __restrict__ boxw, const float* __restrict__ boxb,
                                              const float* __restrict__ boxes,
                                              float* __restrict__ out) {
  int n = blockIdx.x;
  __shared__ float sr[HID];
  __shared__ float sl[21];
  int t = threadIdx.x;
  for (int i = t; i < HID; i += 32) sr[i] = (float)r2[(size_t)n * HID + i];
  __syncthreads();
  if (t < 21) {
    const float* wv = (t < NCLS) ? (clsw + (size_t)t * HID) : (boxw + (size_t)(t - NCLS) * HID);
    float s = 0.f;
    for (int k = 0; k < HID; ++k) s += sr[k] * wv[k];
    s += (t < NCLS) ? clsb[t] : boxb[t - NCLS];
    sl[t] = s;
  }
  __syncthreads();
  if (t == 0) {
    float mx = sl[0];
    for (int i = 1; i < NCLS; ++i) mx = fmaxf(mx, sl[i]);
    float e[NCLS], sum = 0.f;
    for (int i = 0; i < NCLS; ++i) { e[i] = __expf(sl[i] - mx); sum += e[i]; }
    float inv = 1.f / sum;
    float p0 = boxes[n*4+0], p1 = boxes[n*4+1], p2 = boxes[n*4+2], p3 = boxes[n*4+3];
    float cx = (p0 + p2) * 0.5f, cy = (p1 + p3) * 0.5f, w_ = p2 - p0, h_ = p3 - p1;
    float d[5];
    for (int i = 0; i < 5; ++i) d[i] = fminf(fmaxf(sl[NCLS + i], -CLAMPV), CLAMPV);
    float* o = out + (size_t)n * 21;
    o[0] = cx + d[0] * w_;
    o[1] = cy + d[1] * h_;
    o[2] = w_ * __expf(d[2]);
    o[3] = h_ * __expf(d[3]);
    o[4] = d[4];
    for (int i = 0; i < NCLS; ++i) o[5 + i] = e[i] * inv;
  }
}

// ---------------------------------------------------------------------------
extern "C" void kernel_launch(void* const* d_in, const int* in_sizes, int n_in,
                              void* d_out, int out_size, void* d_ws, size_t ws_size,
                              hipStream_t stream) {
  const float* features  = (const float*)d_in[0];
  const float* proposals = (const float*)d_in[1];
  const int*   batch_idx = (const int*)  d_in[2];
  const float* conv_w    = (const float*)d_in[3];
  const float* conv_b    = (const float*)d_in[4];
  const float* align_w   = (const float*)d_in[5];
  const float* align_b   = (const float*)d_in[6];
  const float* fc1_w     = (const float*)d_in[7];
  const float* fc1_b     = (const float*)d_in[8];
  const float* fc2_w     = (const float*)d_in[9];
  const float* fc2_b     = (const float*)d_in[10];
  const float* cls_w     = (const float*)d_in[11];
  const float* cls_b     = (const float*)d_in[12];
  const float* box_w     = (const float*)d_in[13];
  const float* box_b     = (const float*)d_in[14];
  (void)in_sizes; (void)n_in; (void)out_size; (void)ws_size;

  char* ws = (char*)d_ws;
  size_t off = 0;
  auto take = [&](size_t bytes) -> char* {
    char* p = ws + off;
    off = (off + bytes + 255) & ~(size_t)255;
    return p;
  };
  bf16_t* boxfeat = (bf16_t*)take((size_t)NPROP * FD * 2);
  bf16_t* aligned = (bf16_t*)take((size_t)NPROP * FD * 2);
  float*  coarse  = (float*) take((size_t)NPROP * 5 * 4);
  bf16_t* r1      = (bf16_t*)take((size_t)NPROP * HID * 2);
  bf16_t* r2      = (bf16_t*)take((size_t)NPROP * HID * 2);
  bf16_t* convWb  = (bf16_t*)take((size_t)CCH * KCONV * 2);
  bf16_t* fc1Wb   = (bf16_t*)take((size_t)HID * FD * 2);
  bf16_t* fc2Wb   = (bf16_t*)take((size_t)HID * HID * 2);

  cvt_convw<<<(CCH * KCONV + 255) / 256, 256, 0, stream>>>(conv_w, convWb);
  cvt_f32_bf16<<<(HID * FD + 255) / 256, 256, 0, stream>>>(fc1_w, fc1Wb, HID * FD);
  cvt_f32_bf16<<<(HID * HID + 255) / 256, 256, 0, stream>>>(fc2_w, fc2Wb, HID * HID);

  roi_align_k<<<NPROP, 256, 0, stream>>>(features, proposals, batch_idx, boxfeat);
  conv_coarse_k<<<NPROP, 512, 0, stream>>>(boxfeat, convWb, conv_b, align_w, align_b, coarse);
  grid_sample_k<<<NPROP, 256, 0, stream>>>(boxfeat, proposals, coarse, aligned);

  gemm_bf16_k<<<dim3(HID / 128, NPROP / 64), 256, 0, stream>>>(aligned, fc1Wb, fc1_b, r1,
                                                               NPROP, FD, HID, 1);
  gemm_bf16_k<<<dim3(HID / 128, NPROP / 64), 256, 0, stream>>>(r1, fc2Wb, fc2_b, r2,
                                                               NPROP, HID, HID, 1);

  heads_k<<<NPROP, 32, 0, stream>>>(r2, cls_w, cls_b, box_w, box_b, proposals, (float*)d_out);
}
